// ConvAttention_2448131359231
// MI455X (gfx1250) — matrix operand order repaired and checked by reading, not compiled, not run on hardware
//
#include <hip/hip_runtime.h>
#include <hip/hip_bf16.h>

typedef __attribute__((ext_vector_type(16))) _Float16 v16h;
typedef __attribute__((ext_vector_type(8)))  _Float16 v8h;
typedef __attribute__((ext_vector_type(8)))  float    v8f;

#define TEMP_C 0.0005f
#define EPS_C  1e-8f
#define NEG_INF (-__builtin_huge_valf())

// ---------------------------------------------------------------------
// Pack conv weights f32 [M][Cin][KW] -> f16 [KW][Mpad][Cpad], zero-padded.
// ---------------------------------------------------------------------
__global__ void pack_w_kernel(const float* __restrict__ W, _Float16* __restrict__ Wp,
                              int M, int Cin, int KW, int Mpad, int Cpad)
{
    long long idx = (long long)blockIdx.x * blockDim.x + threadIdx.x;
    long long total = (long long)KW * Mpad * Cpad;
    if (idx >= total) return;
    int c  = (int)(idx % Cpad);
    int m  = (int)((idx / Cpad) % Mpad);
    int dk = (int)(idx / ((long long)Cpad * Mpad));
    float v = (m < M && c < Cin) ? W[((size_t)m * Cin + c) * KW + dk] : 0.0f;
    Wp[idx] = (_Float16)v;
}

__global__ void pack_b_kernel(const float* __restrict__ bias, float* __restrict__ bp,
                              int M, int Mpad)
{
    int m = blockIdx.x * blockDim.x + threadIdx.x;
    if (m >= Mpad) return;
    bp[m] = (m < M) ? bias[m] : 0.0f;
}

// ---------------------------------------------------------------------
// Transpose+convert input: f32 [B][C][T] -> f16 [B][T+2][Cpad]
// (rows 0 and T+1 are zero halo for pad=1 convs; channel pad zeroed)
// ---------------------------------------------------------------------
__global__ void transpose_in_kernel(const float* __restrict__ X, _Float16* __restrict__ Xt,
                                    int C, int T, int Cpad)
{
    long long idx = (long long)blockIdx.x * blockDim.x + threadIdx.x;
    long long per_b = (long long)(T + 2) * Cpad;
    if (idx >= per_b) return;
    int b = blockIdx.y;
    int c = (int)(idx % Cpad);
    int t = (int)(idx / Cpad);          // 0..T+1, actual time = t-1
    float v = (t >= 1 && t <= T && c < C) ? X[((size_t)b * C + c) * T + (t - 1)] : 0.0f;
    Xt[(size_t)b * per_b + idx] = (_Float16)v;
}

// ---------------------------------------------------------------------
// Fused conv-as-GEMM, branch-free WMMA:
//   Y[b][t][m] = relu?( sum_dk sum_c Wp[dk][m][c] * Xt[b][t+dk][c] + bias[m] )
// Wp: f16 [KW][Mpad][Kpad]; Xt: f16 [B][T+KW-1][Kpad]; Y: f16 [B][T][Mpad]
// block = (32, 4): 4 waves, each owns one 16x16 output tile.
// ---------------------------------------------------------------------
__global__ void __launch_bounds__(128)
gemm_wmma_kernel(const _Float16* __restrict__ Wp, const _Float16* __restrict__ Xt,
                 const float* __restrict__ biasp, _Float16* __restrict__ Y,
                 int Mpad, int Kpad, int T, int KW, int relu)
{
    const int lane = threadIdx.x;
    const int hi   = lane >> 4;
    const int nTiles = T / 16;
    const int mTiles = Mpad / 16;
    const int tid = blockIdx.x * blockDim.y + threadIdx.y;
    if (tid >= nTiles * mTiles) return;          // wave-uniform exit
    const int n0 = (tid % nTiles) * 16;
    const int m0 = (tid / nTiles) * 16;
    const int b  = blockIdx.z;
    const int rows = T + KW - 1;

    const int am = m0 + (lane & 15);
    const int bn = n0 + (lane & 15);

    v8f acc = {};
    for (int dk = 0; dk < KW; ++dk) {
        const _Float16* wrow = Wp + ((size_t)dk * Mpad + am) * Kpad;
        const _Float16* xrow = Xt + ((size_t)b * rows + (bn + dk)) * Kpad;
        for (int k0 = 0; k0 < Kpad; k0 += 32) {
            v8h a0 = *(const v8h*)(wrow + k0 + hi * 16);
            v8h a1 = *(const v8h*)(wrow + k0 + hi * 16 + 8);
            v16h af = __builtin_shufflevector(a0, a1, 0, 1, 2, 3, 4, 5, 6, 7,
                                              8, 9, 10, 11, 12, 13, 14, 15);
            v16h bf = *(const v16h*)(xrow + k0 + hi * 16);
            acc = __builtin_amdgcn_wmma_f32_16x16x32_f16(false, af, false, bf,
                                                         (short)0, acc, false, false);
        }
    }
    v8f bv = *(const v8f*)(biasp + m0 + hi * 8);
    v8h o;
#pragma unroll
    for (int r = 0; r < 8; ++r) {
        float v = acc[r] + bv[r];
        if (relu) v = fmaxf(v, 0.0f);
        o[r] = (_Float16)v;
    }
    *(v8h*)(Y + ((size_t)b * T + bn) * Mpad + m0 + hi * 8) = o;
}

// ---------------------------------------------------------------------
// out[b][t] = sum_c X[b][t][c]^2   (X f16 [B][T][Cpad], pad cols are zero)
// ---------------------------------------------------------------------
__global__ void sumsq_kernel(const _Float16* __restrict__ X, float* __restrict__ out,
                             int Cpad, int T)
{
    int t = blockIdx.x * blockDim.x + threadIdx.x;
    int b = blockIdx.y;
    if (t >= T) return;
    const _Float16* row = X + ((size_t)b * T + t) * Cpad;
    float s = 0.0f;
    for (int c = 0; c < Cpad; ++c) {
        float v = (float)row[c];
        s += v * v;
    }
    out[(size_t)b * T + t] = s;
}

// ---------------------------------------------------------------------
// Fused attention: qk (WMMA, K=Cpad) -> dist -> log_softmax + log(prior)
// -> mask -> softmax.  One wave per [16 x T2] row block staged in LDS.
// Qp: f16 [B][T1][Cpad], Kp: f16 [B][T2][Cpad]
// ---------------------------------------------------------------------
#define T2C 400
#define T2P (T2C + 8)
#define CPAD_ATT 96
__global__ void __launch_bounds__(32)
attn_wmma_kernel(const _Float16* __restrict__ Qp, const _Float16* __restrict__ Kp,
                 const float* __restrict__ q2, const float* __restrict__ k2,
                 const float* __restrict__ prior, const unsigned char* __restrict__ mask,
                 float* __restrict__ attn, float* __restrict__ logp, int T1)
{
    __shared__ float tile[16][T2P];
    const int lane = threadIdx.x;
    const int hi   = lane >> 4;
    const int t10  = blockIdx.x * 16;
    const int b    = blockIdx.y;

    const _Float16* qrow = Qp + ((size_t)b * T1 + (t10 + (lane & 15))) * CPAD_ATT;

    // Phase 1: x = -TEMP * (|q|^2 + |k|^2 - 2 q.k) into LDS
    for (int n0 = 0; n0 < T2C; n0 += 16) {
        const _Float16* krow = Kp + ((size_t)b * T2C + (n0 + (lane & 15))) * CPAD_ATT;
        v8f acc = {};
#pragma unroll
        for (int k0 = 0; k0 < CPAD_ATT; k0 += 32) {
            v8h a0 = *(const v8h*)(qrow + k0 + hi * 16);
            v8h a1 = *(const v8h*)(qrow + k0 + hi * 16 + 8);
            v16h af = __builtin_shufflevector(a0, a1, 0, 1, 2, 3, 4, 5, 6, 7,
                                              8, 9, 10, 11, 12, 13, 14, 15);
            v16h bf = *(const v16h*)(krow + k0 + hi * 16);
            acc = __builtin_amdgcn_wmma_f32_16x16x32_f16(false, af, false, bf,
                                                         (short)0, acc, false, false);
        }
#pragma unroll
        for (int r = 0; r < 8; ++r) {
            int m  = r + hi * 8;
            int t2 = n0 + (lane & 15);
            float dist = q2[(size_t)b * T1 + t10 + m] + k2[(size_t)b * T2C + t2]
                       - 2.0f * acc[r];
            tile[m][t2] = -TEMP_C * dist;
        }
    }
    __syncthreads();

    // Phase 2: per-row softmaxes (lane-parallel over T2, shfl reductions, wave32)
    for (int r = 0; r < 16; ++r) {
        const size_t rowoff = ((size_t)b * T1 + (t10 + r)) * T2C;
        float mx = NEG_INF;
        for (int t2 = lane; t2 < T2C; t2 += 32) mx = fmaxf(mx, tile[r][t2]);
        for (int i = 16; i > 0; i >>= 1) mx = fmaxf(mx, __shfl_xor(mx, i, 32));
        float s = 0.0f;
        for (int t2 = lane; t2 < T2C; t2 += 32) s += __expf(tile[r][t2] - mx);
        for (int i = 16; i > 0; i >>= 1) s += __shfl_xor(s, i, 32);
        const float logZ = mx + __logf(s);

        float mx2 = NEG_INF;
        for (int t2 = lane; t2 < T2C; t2 += 32) {
            float y = tile[r][t2] - logZ + __logf(prior[rowoff + t2] + EPS_C);
            logp[rowoff + t2] = y;
            float ym = mask[(size_t)b * T2C + t2] ? NEG_INF : y;
            tile[r][t2] = ym;
            mx2 = fmaxf(mx2, ym);
        }
        for (int i = 16; i > 0; i >>= 1) mx2 = fmaxf(mx2, __shfl_xor(mx2, i, 32));
        float s2 = 0.0f;
        for (int t2 = lane; t2 < T2C; t2 += 32) {
            float e = __expf(tile[r][t2] - mx2);
            tile[r][t2] = e;
            s2 += e;
        }
        for (int i = 16; i > 0; i >>= 1) s2 += __shfl_xor(s2, i, 32);
        const float inv = 1.0f / s2;
        for (int t2 = lane; t2 < T2C; t2 += 32)
            attn[rowoff + t2] = tile[r][t2] * inv;
    }
}

// ---------------------------------------------------------------------
extern "C" void kernel_launch(void* const* d_in, const int* in_sizes, int n_in,
                              void* d_out, int out_size, void* d_ws, size_t ws_size,
                              hipStream_t stream) {
    (void)in_sizes; (void)n_in; (void)out_size; (void)ws_size;
    const int B = 32, T1 = 2000, T2 = 400;

    const float* queries = (const float*)d_in[0];              // [B,80,2000]
    const float* keys    = (const float*)d_in[1];              // [B,512,400]
    /* d_in[2] = query_lens (unused by reference output) */
    const unsigned char* mask = (const unsigned char*)d_in[3]; // [B,400,1] bool
    const float* prior   = (const float*)d_in[4];              // [B,2000,400]
    const float* wk1 = (const float*)d_in[5];  const float* bk1 = (const float*)d_in[6];
    const float* wk2 = (const float*)d_in[7];  const float* bk2 = (const float*)d_in[8];
    const float* wq1 = (const float*)d_in[9];  const float* bq1 = (const float*)d_in[10];
    const float* wq2 = (const float*)d_in[11]; const float* bq2 = (const float*)d_in[12];
    const float* wq3 = (const float*)d_in[13]; const float* bq3 = (const float*)d_in[14];

    float* attn_out = (float*)d_out;                           // [B,1,T1,T2]
    float* logp_out = attn_out + (size_t)B * T1 * T2;          // [B,1,T1,T2]

    // ---- workspace carve-up ----
    char* ws = (char*)d_ws;
    size_t off = 0;
    auto alloc = [&](size_t bytes) -> void* {
        void* p = ws + off;
        off += (bytes + 255) & ~(size_t)255;
        return p;
    };
    _Float16* Kt    = (_Float16*)alloc((size_t)B * 402  * 512 * 2);  // keys^T + halo
    _Float16* Qt    = (_Float16*)alloc((size_t)B * 2002 * 96  * 2);  // queries^T + halo
    _Float16* K1    = (_Float16*)alloc((size_t)B * 400  * 1024 * 2); // [B][T2][1024]
    _Float16* Q1    = (_Float16*)alloc((size_t)B * 2000 * 160 * 2);  // [B][T1][160]
    _Float16* Q2    = (_Float16*)alloc((size_t)B * 2000 * 96  * 2);  // [B][T1][96]
    _Float16* Qenc  = (_Float16*)alloc((size_t)B * 2000 * 96  * 2);  // [B][T1][96]
    _Float16* Kenc  = (_Float16*)alloc((size_t)B * 400  * 96  * 2);  // [B][T2][96]
    _Float16* wk1p  = (_Float16*)alloc((size_t)3 * 1024 * 512 * 2);
    _Float16* wk2p  = (_Float16*)alloc((size_t)1 * 96 * 1024 * 2);
    _Float16* wq1p  = (_Float16*)alloc((size_t)3 * 160 * 96 * 2);
    _Float16* wq2p  = (_Float16*)alloc((size_t)1 * 96 * 160 * 2);
    _Float16* wq3p  = (_Float16*)alloc((size_t)1 * 96 * 96 * 2);
    float* bk1p = (float*)alloc(1024 * 4);
    float* bk2p = (float*)alloc(96 * 4);
    float* bq1p = (float*)alloc(160 * 4);
    float* bq2p = (float*)alloc(96 * 4);
    float* bq3p = (float*)alloc(96 * 4);
    float* q2   = (float*)alloc((size_t)B * T1 * 4);
    float* k2   = (float*)alloc((size_t)B * T2 * 4);

    // ---- pack weights / biases (one-shot, tiny) ----
    auto gsz = [](long long n) { return (unsigned)((n + 255) / 256); };
    pack_w_kernel<<<gsz(3LL*1024*512), 256, 0, stream>>>(wk1, wk1p, 1024, 512, 3, 1024, 512);
    pack_w_kernel<<<gsz(1LL*96*1024),  256, 0, stream>>>(wk2, wk2p,   80, 1024, 1,   96, 1024);
    pack_w_kernel<<<gsz(3LL*160*96),   256, 0, stream>>>(wq1, wq1p,  160,   80, 3,  160,   96);
    pack_w_kernel<<<gsz(1LL*96*160),   256, 0, stream>>>(wq2, wq2p,   80,  160, 1,   96,  160);
    pack_w_kernel<<<gsz(1LL*96*96),    256, 0, stream>>>(wq3, wq3p,   80,   80, 1,   96,   96);
    pack_b_kernel<<<4, 256, 0, stream>>>(bk1, bk1p, 1024, 1024);
    pack_b_kernel<<<1, 96,  0, stream>>>(bk2, bk2p,   80,   96);
    pack_b_kernel<<<1, 160, 0, stream>>>(bq1, bq1p,  160,  160);
    pack_b_kernel<<<1, 96,  0, stream>>>(bq2, bq2p,   80,   96);
    pack_b_kernel<<<1, 96,  0, stream>>>(bq3, bq3p,   80,   96);

    // ---- transpose inputs to [B][T+2][Cpad] f16 with halo ----
    transpose_in_kernel<<<dim3(gsz((long long)402 * 512), B), 256, 0, stream>>>(
        keys, Kt, 512, T2, 512);
    transpose_in_kernel<<<dim3(gsz((long long)2002 * 96), B), 256, 0, stream>>>(
        queries, Qt, 80, T1, 96);

    // ---- key_proj ----
    // conv(512->1024,k3)+relu : tiles = 25*64 = 1600
    gemm_wmma_kernel<<<dim3((1600 + 3) / 4, 1, B), dim3(32, 4), 0, stream>>>(
        wk1p, Kt, bk1p, K1, 1024, 512, T2, 3, 1);
    // conv(1024->80,k1) : tiles = 25*6 = 150
    gemm_wmma_kernel<<<dim3((150 + 3) / 4, 1, B), dim3(32, 4), 0, stream>>>(
        wk2p, K1, bk2p, Kenc, 96, 1024, T2, 1, 0);

    // ---- query_proj ----
    // conv(80->160,k3)+relu : tiles = 125*10 = 1250
    gemm_wmma_kernel<<<dim3((1250 + 3) / 4, 1, B), dim3(32, 4), 0, stream>>>(
        wq1p, Qt, bq1p, Q1, 160, 96, T1, 3, 1);
    // conv(160->80,k1)+relu : tiles = 125*6 = 750
    gemm_wmma_kernel<<<dim3((750 + 3) / 4, 1, B), dim3(32, 4), 0, stream>>>(
        wq2p, Q1, bq2p, Q2, 96, 160, T1, 1, 1);
    // conv(80->80,k1) : tiles = 750
    gemm_wmma_kernel<<<dim3((750 + 3) / 4, 1, B), dim3(32, 4), 0, stream>>>(
        wq3p, Q2, bq3p, Qenc, 96, 96, T1, 1, 0);

    // ---- |q|^2, |k|^2 ----
    sumsq_kernel<<<dim3((T1 + 255) / 256, B), 256, 0, stream>>>(Qenc, q2, 96, T1);
    sumsq_kernel<<<dim3((T2 + 255) / 256, B), 256, 0, stream>>>(Kenc, k2, 96, T2);

    // ---- fused qk + dist + log_softmax + prior + mask + softmax ----
    attn_wmma_kernel<<<dim3(T1 / 16, B), 32, 0, stream>>>(
        Qenc, Kenc, q2, k2, prior, mask, attn_out, logp_out, T1);
}
